// SynthesizerLayer_22737556865754
// MI455X (gfx1250) — matrix operand order repaired and checked by reading, not compiled, not run on hardware
//
#include <hip/hip_runtime.h>
#include <hip/hip_fp16.h>
#include <math.h>

typedef __attribute__((ext_vector_type(16))) _Float16 v16h;
typedef __attribute__((ext_vector_type(8)))  _Float16 v8h;
typedef __attribute__((ext_vector_type(8)))  float    v8f;
typedef __attribute__((ext_vector_type(4)))  float    v4f;

#define B_  8
#define N_  512
#define M_  512
#define D_  512
#define F_  512
#define H_  8
#define DH  64
#define FH  64

// ---------------------------------------------------------------------------
// k1: Q1[h][n][b16][d] = gelu( X[b,n,h*64+d] * diag(W_Q_1)[h,n,d] ), f16.
// X / W_Q_1 are single-pass streams -> non-temporal loads (spare L2 for Q1/A).
// ---------------------------------------------------------------------------
__global__ __launch_bounds__(256) void k_q1(const float* __restrict__ X,
                                            const float* __restrict__ WQ1,
                                            _Float16* __restrict__ Q1) {
  int idx = blockIdx.x * 256 + threadIdx.x;   // [h][n][b16][d]
  int d = idx & 63;
  int b = (idx >> 6) & 15;
  int n = (idx >> 10) & 511;
  int h = idx >> 19;
  _Float16 out = (_Float16)0.0f;
  if (b < B_) {
    float x = __builtin_nontemporal_load(X + ((size_t)(b * N_ + n)) * D_ + h * DH + d);
    float w = __builtin_nontemporal_load(WQ1 + (((size_t)(h * N_ + n)) * DH + d) * DH + d);
    float v = x * w;
    float g = 0.5f * v * (1.0f + erff(v * 0.70710678118654752f));  // exact gelu
    out = (_Float16)g;
  }
  Q1[idx] = out;
}

// ---------------------------------------------------------------------------
// k1b: Vs[b][h][f][m] = V[b,m,h*64+f] * diag(W_V)[h,f], f16, transposed via LDS
// so k3's WMMA B-operands are contiguous 32B v16h loads along m.
// V is a single-pass stream -> non-temporal.
// ---------------------------------------------------------------------------
__global__ __launch_bounds__(256) void k_vs(const float* __restrict__ V,
                                            const float* __restrict__ WV,
                                            _Float16* __restrict__ Vs) {
  int blk = blockIdx.x;            // [bh][mt(16)][ft(2)]
  int ft = blk & 1;
  int mt = (blk >> 1) & 15;
  int bh = blk >> 5;
  int b = bh >> 3, h = bh & 7;
  __shared__ float tile[32][33];
  int t = threadIdx.x;
  int fi = t & 31, mr = t >> 5;    // read: f fastest (coalesced in V)
#pragma unroll
  for (int i = 0; i < 4; i++) {
    int m = mt * 32 + mr + 8 * i;
    tile[mr + 8 * i][fi] =
        __builtin_nontemporal_load(V + ((size_t)b * M_ + m) * F_ + h * FH + ft * 32 + fi);
  }
  __syncthreads();
  int mo = t & 31, fr = t >> 5;    // write: m fastest (coalesced in Vs)
#pragma unroll
  for (int i = 0; i < 4; i++) {
    int f = ft * 32 + fr + 8 * i;
    float wv = WV[((size_t)(h * FH + f)) * FH + f];
    Vs[(((size_t)bh) * FH + f) * M_ + mt * 32 + mo] = (_Float16)(tile[mo][fr + 8 * i] * wv);
  }
}

// ---------------------------------------------------------------------------
// k2: per (h,n): logits L[16 x 512] = Q1tile(16x64) x W_Q_2(h,n)(64x512 ^T)
// via v_wmma_f32_16x16x32_f16. W_Q_2 (537 MB, touched exactly once) is
// streamed with NT hints so it doesn't wash the reused tensors out of L2.
// Then softmax over m in LDS, A[b,h,n,m] written f16.
// ---------------------------------------------------------------------------
#define LSTR 520
__global__ __launch_bounds__(128) void k_logits(const _Float16* __restrict__ Q1,
                                                const float* __restrict__ WQ2,
                                                _Float16* __restrict__ A) {
  int hn = blockIdx.x;             // h*N + n
  int n = hn & 511, h = hn >> 9;
  __shared__ float L[16][LSTR];
  __shared__ float red[8][16];
  __shared__ float rowstat[8];
  int t = threadIdx.x;
  int w = t >> 5, lane = t & 31;
  int r = lane & 15, kh = lane >> 4;

  // A operands (reused for all 8 m-tiles this wave handles)
  const _Float16* q1p = Q1 + (((size_t)hn * 16 + r) * DH);
  v16h a0, a1;
  {
    v8h lo = *(const v8h*)(q1p + kh * 16);
    v8h hi = *(const v8h*)(q1p + kh * 16 + 8);
#pragma unroll
    for (int i = 0; i < 8; i++) { a0[i] = lo[i]; a0[i + 8] = hi[i]; }
    lo = *(const v8h*)(q1p + 32 + kh * 16);
    hi = *(const v8h*)(q1p + 32 + kh * 16 + 8);
#pragma unroll
    for (int i = 0; i < 8; i++) { a1[i] = lo[i]; a1[i + 8] = hi[i]; }
  }

#pragma unroll 2
  for (int j = 0; j < 8; j++) {
    int mt = w * 8 + j;
    int m = mt * 16 + r;                                // B column = lane&15
    const float* bp = WQ2 + ((size_t)hn * M_ + m) * DH; // 64 contiguous f32
    v8f acc = {};
    {
      v16h bm;
      const v4f* p4 = (const v4f*)(bp + kh * 16);
#pragma unroll
      for (int q = 0; q < 4; q++) { v4f x = __builtin_nontemporal_load(p4 + q);
#pragma unroll
        for (int i = 0; i < 4; i++) bm[q * 4 + i] = (_Float16)x[i]; }
      acc = __builtin_amdgcn_wmma_f32_16x16x32_f16(false, a0, false, bm, (short)0, acc, false, false);
    }
    {
      v16h bm;
      const v4f* p4 = (const v4f*)(bp + 32 + kh * 16);
#pragma unroll
      for (int q = 0; q < 4; q++) { v4f x = __builtin_nontemporal_load(p4 + q);
#pragma unroll
        for (int i = 0; i < 4; i++) bm[q * 4 + i] = (_Float16)x[i]; }
      acc = __builtin_amdgcn_wmma_f32_16x16x32_f16(false, a1, false, bm, (short)0, acc, false, false);
    }
#pragma unroll
    for (int v = 0; v < 8; v++) L[kh * 8 + v][mt * 16 + r] = acc[v];
  }
  __syncthreads();

  // softmax over m for rows b=0..7 (rows 8..15 are padding, ignored)
  int rr = t >> 4;       // 0..7
  int s  = t & 15;
  float pmax = -1e30f;
#pragma unroll 4
  for (int i = 0; i < 32; i++) pmax = fmaxf(pmax, L[rr][s + 16 * i]);
  red[rr][s] = pmax;
  __syncthreads();
  if (s == 0) {
    float mx = red[rr][0];
#pragma unroll
    for (int i = 1; i < 16; i++) mx = fmaxf(mx, red[rr][i]);
    rowstat[rr] = mx;
  }
  __syncthreads();
  float mx = rowstat[rr];
  float psum = 0.0f;
#pragma unroll 4
  for (int i = 0; i < 32; i++) {
    int m = s + 16 * i;
    float e = expf(L[rr][m] - mx);
    L[rr][m] = e;
    psum += e;
  }
  red[rr][s] = psum;
  __syncthreads();
  if (s == 0) {
    float sm = 0.0f;
#pragma unroll
    for (int i = 0; i < 16; i++) sm += red[rr][i];
    rowstat[rr] = sm;
  }
  __syncthreads();
  float inv = 1.0f / rowstat[rr];
  _Float16* ap = A + (((size_t)(rr * H_ + h)) * N_ + n) * M_;
#pragma unroll 4
  for (int i = 0; i < 32; i++) {
    int m = s + 16 * i;
    ap[m] = (_Float16)(L[rr][m] * inv);
  }
}

// ---------------------------------------------------------------------------
// k3: per (b,h): Y[n, h*64+f] = A(512x512) @ Vs(512x64) via WMMA.
// Wave = 16 rows x 64 cols, 16 k-steps. Vs B-tiles are contiguous v16h loads.
// ---------------------------------------------------------------------------
__global__ __launch_bounds__(128) void k_av(const _Float16* __restrict__ A,
                                            const _Float16* __restrict__ Vs,
                                            _Float16* __restrict__ Y) {
  int blk = blockIdx.x;            // bh*8 + ntile
  int nt = blk & 7;
  int bh = blk >> 3;
  int b = bh >> 3, h = bh & 7;
  int t = threadIdx.x;
  int w = t >> 5, lane = t & 31;
  int r = lane & 15, kh = lane >> 4;
  int n = nt * 64 + w * 16 + r;
  const _Float16* arow = A + (((size_t)bh) * N_ + n) * M_;
  const _Float16* vb = Vs + ((size_t)bh * FH) * M_;
  v8f acc0 = {}, acc1 = {}, acc2 = {}, acc3 = {};
#pragma unroll 4
  for (int ks = 0; ks < 16; ks++) {
    int k0 = ks * 32;
    v16h am;
    v8h lo = *(const v8h*)(arow + k0 + kh * 16);
    v8h hi = *(const v8h*)(arow + k0 + kh * 16 + 8);
#pragma unroll
    for (int i = 0; i < 8; i++) { am[i] = lo[i]; am[i + 8] = hi[i]; }
    v16h b0 = *(const v16h*)(vb + ((size_t)(0 * 16 + r)) * M_ + k0 + kh * 16);
    v16h b1 = *(const v16h*)(vb + ((size_t)(1 * 16 + r)) * M_ + k0 + kh * 16);
    v16h b2 = *(const v16h*)(vb + ((size_t)(2 * 16 + r)) * M_ + k0 + kh * 16);
    v16h b3 = *(const v16h*)(vb + ((size_t)(3 * 16 + r)) * M_ + k0 + kh * 16);
    acc0 = __builtin_amdgcn_wmma_f32_16x16x32_f16(false, am, false, b0, (short)0, acc0, false, false);
    acc1 = __builtin_amdgcn_wmma_f32_16x16x32_f16(false, am, false, b1, (short)0, acc1, false, false);
    acc2 = __builtin_amdgcn_wmma_f32_16x16x32_f16(false, am, false, b2, (short)0, acc2, false, false);
    acc3 = __builtin_amdgcn_wmma_f32_16x16x32_f16(false, am, false, b3, (short)0, acc3, false, false);
  }
#pragma unroll
  for (int v = 0; v < 8; v++) {
    int nrow = nt * 64 + w * 16 + kh * 8 + v;
    _Float16* yp = Y + ((size_t)b * N_ + nrow) * F_ + h * FH;
    yp[ 0 + r] = (_Float16)acc0[v];
    yp[16 + r] = (_Float16)acc1[v];
    yp[32 + r] = (_Float16)acc2[v];
    yp[48 + r] = (_Float16)acc3[v];
  }
}

// ---------------------------------------------------------------------------
// k4: out[p, j] = Y[p,:] . O_w[j,:] + O_b[j]   (p = b*N+n), f32 output.
// O_w is reused across all p-tiles -> keep default (cacheable) hints.
// ---------------------------------------------------------------------------
__global__ __launch_bounds__(128) void k_proj(const _Float16* __restrict__ Y,
                                              const float* __restrict__ Ow,
                                              const float* __restrict__ Ob,
                                              float* __restrict__ out) {
  int blk = blockIdx.x;            // ptile(64)*8 + jtile(8)
  int jt = blk & 7;
  int pt = blk >> 3;
  int t = threadIdx.x;
  int w = t >> 5, lane = t & 31;
  int r = lane & 15, kh = lane >> 4;
  int p = pt * 64 + w * 16 + r;
  const _Float16* arow = Y + (size_t)p * F_;
  v8f acc0 = {}, acc1 = {}, acc2 = {}, acc3 = {};
#pragma unroll 2
  for (int ks = 0; ks < 16; ks++) {
    int k0 = ks * 32;
    v16h am;
    v8h lo = *(const v8h*)(arow + k0 + kh * 16);
    v8h hi = *(const v8h*)(arow + k0 + kh * 16 + 8);
#pragma unroll
    for (int i = 0; i < 8; i++) { am[i] = lo[i]; am[i + 8] = hi[i]; }
#pragma unroll
    for (int ct = 0; ct < 4; ct++) {
      int j = jt * 64 + ct * 16 + r;                 // B column = lane&15
      const v4f* p4 = (const v4f*)(Ow + (size_t)j * F_ + k0 + kh * 16);
      v16h bm;
#pragma unroll
      for (int q = 0; q < 4; q++) { v4f x = p4[q];
#pragma unroll
        for (int i = 0; i < 4; i++) bm[q * 4 + i] = (_Float16)x[i]; }
      switch (ct) {
        case 0: acc0 = __builtin_amdgcn_wmma_f32_16x16x32_f16(false, am, false, bm, (short)0, acc0, false, false); break;
        case 1: acc1 = __builtin_amdgcn_wmma_f32_16x16x32_f16(false, am, false, bm, (short)0, acc1, false, false); break;
        case 2: acc2 = __builtin_amdgcn_wmma_f32_16x16x32_f16(false, am, false, bm, (short)0, acc2, false, false); break;
        default: acc3 = __builtin_amdgcn_wmma_f32_16x16x32_f16(false, am, false, bm, (short)0, acc3, false, false); break;
      }
    }
  }
#pragma unroll
  for (int v = 0; v < 8; v++) {
    int prow = pt * 64 + w * 16 + kh * 8 + v;
    float* op = out + (size_t)prow * F_ + jt * 64;
    op[ 0 + r] = acc0[v] + Ob[jt * 64 +  0 + r];
    op[16 + r] = acc1[v] + Ob[jt * 64 + 16 + r];
    op[32 + r] = acc2[v] + Ob[jt * 64 + 32 + r];
    op[48 + r] = acc3[v] + Ob[jt * 64 + 48 + r];
  }
}

// ---------------------------------------------------------------------------
extern "C" void kernel_launch(void* const* d_in, const int* in_sizes, int n_in,
                              void* d_out, int out_size, void* d_ws, size_t ws_size,
                              hipStream_t stream) {
  (void)in_sizes; (void)n_in; (void)out_size; (void)ws_size;
  const float* X   = (const float*)d_in[0];
  const float* V   = (const float*)d_in[1];
  const float* WQ1 = (const float*)d_in[2];
  const float* WQ2 = (const float*)d_in[3];
  const float* WV  = (const float*)d_in[4];
  const float* Ow  = (const float*)d_in[5];
  const float* Ob  = (const float*)d_in[6];
  float* out = (float*)d_out;

  char* ws = (char*)d_ws;
  _Float16* Q1 = (_Float16*)(ws);                         //  8,388,608 B
  _Float16* Vs = (_Float16*)(ws + (size_t)8388608);       // 33,554,432 B
  _Float16* A  = (_Float16*)(ws + (size_t)41943040);      // 33,554,432 B
  _Float16* Y  = (_Float16*)(ws + (size_t)75497472);      //  4,194,304 B

  k_q1   <<<16384, 256, 0, stream>>>(X, WQ1, Q1);
  k_vs   <<< 2048, 256, 0, stream>>>(V, WV, Vs);
  k_logits<<<4096, 128, 0, stream>>>(Q1, WQ2, A);
  k_av   <<<  512, 128, 0, stream>>>(A, Vs, Y);
  k_proj <<<  512, 128, 0, stream>>>(Y, Ow, Ob, out);
}
